// SelfAttention_68702296867381
// MI455X (gfx1250) — matrix operand order repaired and checked by reading, not compiled, not run on hardware
//
#include <hip/hip_runtime.h>
#include <hip/hip_bf16.h>

typedef __bf16 bf16;
typedef __attribute__((ext_vector_type(16))) __bf16 v16bf;
typedef __attribute__((ext_vector_type(8)))  __bf16 v8bf;
typedef __attribute__((ext_vector_type(4)))  __bf16 v4bf;
typedef __attribute__((ext_vector_type(8)))  float  v8f;
typedef int v4i __attribute__((vector_size(16)));

#define NROWS 8192
#define DDIM  2048

// Async global->LDS path (gfx1250): guarded so host pass / older toolchains
// fall back to the proven synchronous staging loop.
// Builtin signature (from hipcc diagnostics): param0 = int4 AS(1)*, so pass
// int4-typed pointers in the proper address spaces.
#if defined(__gfx1250__) &&                                        \
    __has_builtin(__builtin_amdgcn_global_load_async_to_lds_b128) && \
    __has_builtin(__builtin_amdgcn_s_wait_asynccnt)
#define USE_ASYNC 1
#define ASYNC_LD_B128(gp, lp)                                      \
  __builtin_amdgcn_global_load_async_to_lds_b128(                  \
      (__attribute__((address_space(1))) v4i*)(const void*)(gp),   \
      (__attribute__((address_space(3))) v4i*)(void*)(lp), 0, 0)
#else
#define USE_ASYNC 0
#endif

// ---------------- fp32 -> bf16 (plain) ----------------
__global__ __launch_bounds__(256) void cvt_f32_bf16(const float* __restrict__ x,
                                                    bf16* __restrict__ y) {
  size_t idx = ((size_t)blockIdx.x * 256 + threadIdx.x) * 4;
  float4 f = *reinterpret_cast<const float4*>(x + idx);
  v4bf o;
  o[0] = (bf16)f.x; o[1] = (bf16)f.y; o[2] = (bf16)f.z; o[3] = (bf16)f.w;
  *reinterpret_cast<v4bf*>(y + idx) = o;
}

// ---------------- fp32 -> bf16 with transpose (for weights: writes W^T) ----
__global__ __launch_bounds__(256) void cvt_f32_bf16_T(const float* __restrict__ x,
                                                      bf16* __restrict__ y, int d) {
  __shared__ float tile[32][33];
  const int tx = threadIdx.x, ty = threadIdx.y;
  const int gc = blockIdx.x * 32 + tx;
  const int gr0 = blockIdx.y * 32;
#pragma unroll
  for (int i = 0; i < 4; i++)
    tile[ty + 8 * i][tx] = x[(size_t)(gr0 + ty + 8 * i) * d + gc];
  __syncthreads();
  const int oc = blockIdx.y * 32 + tx;
  const int or0 = blockIdx.x * 32;
#pragma unroll
  for (int i = 0; i < 4; i++)
    y[(size_t)(or0 + ty + 8 * i) * d + oc] = (bf16)tile[tx][ty + 8 * i];
}

// ---------------- row softmax, output transposed bf16 ----------------------
// Y^T[j][row] = softmax_row(X[row])[j]   (softmax over axis=1 / columns)
__global__ __launch_bounds__(256) void softmax_rows_T(const float* __restrict__ x,
                                                      bf16* __restrict__ yT, int d) {
  __shared__ float red[256];
  const int row = blockIdx.x;
  const int tid = threadIdx.x;
  float vals[8];
  float mx = -3.402823466e38f;
#pragma unroll
  for (int i = 0; i < 8; i++) {
    vals[i] = x[(size_t)row * d + tid + i * 256];
    mx = fmaxf(mx, vals[i]);
  }
  red[tid] = mx; __syncthreads();
  for (int s = 128; s > 0; s >>= 1) {
    if (tid < s) red[tid] = fmaxf(red[tid], red[tid + s]);
    __syncthreads();
  }
  mx = red[0]; __syncthreads();
  float sum = 0.f;
#pragma unroll
  for (int i = 0; i < 8; i++) { vals[i] = __expf(vals[i] - mx); sum += vals[i]; }
  red[tid] = sum; __syncthreads();
  for (int s = 128; s > 0; s >>= 1) {
    if (tid < s) red[tid] += red[tid + s];
    __syncthreads();
  }
  const float inv = 1.0f / red[0];
#pragma unroll
  for (int i = 0; i < 8; i++)
    yT[(size_t)(tid + i * 256) * d + row] = (bf16)(vals[i] * inv);
}

// ---------------- WMMA GEMM:  C[M,N] = A[M,K] * Bt[N,K]^T -------------------
// A row-major (lda=K-stride), Bt row-major (B transposed, ldb=K-stride).
// 256 threads = 8 wave32; block tile 128x128, K-step 32.
// Each wave: 32x64 sub-tile = 2x4 accumulators of v_wmma_f32_16x16x32_bf16.
// USE_ASYNC: double-buffered LDS filled by global_load_async_to_lds_b128,
// synchronized with s_wait_asynccnt (in-order completion per wave) + barrier.
constexpr int BM = 128, BN = 128, BK = 32, LDSB = BK + 8;  // 40 bf16 = 80B row

template <typename OutT, bool TRANS>
__global__ __launch_bounds__(256) void gemm_bf16_wmma(
    const bf16* __restrict__ A, const bf16* __restrict__ Bt, OutT* __restrict__ C,
    int M, int N, int K, int lda, int ldb, int ldc) {
#if USE_ASYNC
  __shared__ bf16 sA[2][BM * LDSB];
  __shared__ bf16 sB[2][BN * LDSB];
#else
  __shared__ bf16 sA[1][BM * LDSB];
  __shared__ bf16 sB[1][BN * LDSB];
#endif

  const int tid  = threadIdx.x;
  const int lane = tid & 31;
  const int wave = tid >> 5;
  const int wm = (wave >> 1) * 32;   // wave row offset in block tile
  const int wn = (wave & 1) * 64;    // wave col offset in block tile
  const int h   = lane >> 4;         // lane half (ISA 16-bit fragment layouts)
  const int l16 = lane & 15;

  const int blockM = blockIdx.y * BM;
  const int blockN = blockIdx.x * BN;

  // Global staging: 512 chunks of 8 bf16 (16B) per tile; 2 chunks per thread.
  const int c0 = tid * 2, c1 = tid * 2 + 1;
  const int r0 = c0 >> 2, o0 = (c0 & 3) * 8;
  const int r1 = c1 >> 2, o1 = (c1 & 3) * 8;
  const bf16* gA0 = A  + (size_t)(blockM + r0) * lda + o0;
  const bf16* gA1 = A  + (size_t)(blockM + r1) * lda + o1;
  const bf16* gB0 = Bt + (size_t)(blockN + r0) * ldb + o0;
  const bf16* gB1 = Bt + (size_t)(blockN + r1) * ldb + o1;

  v8f acc[2][4];
  const v8f vzero = {0.f, 0.f, 0.f, 0.f, 0.f, 0.f, 0.f, 0.f};
#pragma unroll
  for (int tm = 0; tm < 2; tm++)
#pragma unroll
    for (int tn = 0; tn < 4; tn++) acc[tm][tn] = vzero;

  int buf = 0;
#if USE_ASYNC
  // Prologue: fill buffer 0 (4 async-load instructions per wave).
  ASYNC_LD_B128(gA0, &sA[0][r0 * LDSB + o0]);
  ASYNC_LD_B128(gA1, &sA[0][r1 * LDSB + o1]);
  ASYNC_LD_B128(gB0, &sB[0][r0 * LDSB + o0]);
  ASYNC_LD_B128(gB1, &sB[0][r1 * LDSB + o1]);
#endif

  for (int kk = 0; kk < K; kk += BK) {
#if USE_ASYNC
    const bool more = (kk + BK) < K;
    if (more) {  // fill the other buffer while we compute on this one
      const int nb = buf ^ 1;
      ASYNC_LD_B128(gA0 + kk + BK, &sA[nb][r0 * LDSB + o0]);
      ASYNC_LD_B128(gA1 + kk + BK, &sA[nb][r1 * LDSB + o1]);
      ASYNC_LD_B128(gB0 + kk + BK, &sB[nb][r0 * LDSB + o0]);
      ASYNC_LD_B128(gB1 + kk + BK, &sB[nb][r1 * LDSB + o1]);
      __builtin_amdgcn_s_wait_asynccnt(4);  // in-order: current buf's 4 done
    } else {
      __builtin_amdgcn_s_wait_asynccnt(0);
    }
    __syncthreads();  // publish LDS writes from all waves
#else
    v8bf la0 = *reinterpret_cast<const v8bf*>(gA0 + kk);
    v8bf la1 = *reinterpret_cast<const v8bf*>(gA1 + kk);
    v8bf lb0 = *reinterpret_cast<const v8bf*>(gB0 + kk);
    v8bf lb1 = *reinterpret_cast<const v8bf*>(gB1 + kk);
    if (kk + BK < K) {  // gfx1250 global_prefetch_b8 for next K-slice
      __builtin_prefetch(gA0 + kk + BK, 0, 1);
      __builtin_prefetch(gB0 + kk + BK, 0, 1);
    }
    __syncthreads();  // previous iteration's LDS reads done
    *reinterpret_cast<v8bf*>(&sA[0][r0 * LDSB + o0]) = la0;
    *reinterpret_cast<v8bf*>(&sA[0][r1 * LDSB + o1]) = la1;
    *reinterpret_cast<v8bf*>(&sB[0][r0 * LDSB + o0]) = lb0;
    *reinterpret_cast<v8bf*>(&sB[0][r1 * LDSB + o1]) = lb1;
    __syncthreads();
#endif

    const bf16* cA = &sA[buf][0];
    const bf16* cB = &sB[buf][0];

    // A fragment: lane holds row (l16), K = {8h..8h+7, 16+8h..23+8h}
    v16bf afr[2];
#pragma unroll
    for (int tm = 0; tm < 2; tm++) {
      const int r = wm + tm * 16 + l16;
      v8bf lo = *reinterpret_cast<const v8bf*>(&cA[r * LDSB + 8 * h]);
      v8bf hi = *reinterpret_cast<const v8bf*>(&cA[r * LDSB + 16 + 8 * h]);
      afr[tm] = __builtin_shufflevector(lo, hi, 0, 1, 2, 3, 4, 5, 6, 7,
                                        8, 9, 10, 11, 12, 13, 14, 15);
    }
    // B fragment: lane holds col (l16), K = {8h..8h+7, 16+8h..23+8h} (Bt row = col)
    v16bf bfr[4];
#pragma unroll
    for (int tn = 0; tn < 4; tn++) {
      const int r = wn + tn * 16 + l16;
      v8bf lo = *reinterpret_cast<const v8bf*>(&cB[r * LDSB + 8 * h]);
      v8bf hi = *reinterpret_cast<const v8bf*>(&cB[r * LDSB + 16 + 8 * h]);
      bfr[tn] = __builtin_shufflevector(lo, hi, 0, 1, 2, 3, 4, 5, 6, 7,
                                        8, 9, 10, 11, 12, 13, 14, 15);
    }
#pragma unroll
    for (int tm = 0; tm < 2; tm++)
#pragma unroll
      for (int tn = 0; tn < 4; tn++)
        acc[tm][tn] = __builtin_amdgcn_wmma_f32_16x16x32_bf16(
            false, afr[tm], false, bfr[tn], (short)0, acc[tm][tn], false, false);

#if USE_ASYNC
    __syncthreads();  // all waves done reading buf before it is refilled
    buf ^= 1;
#endif
  }

  // Epilogue. D layout: element j -> row 8h+j, col l16 (per 32-bit C/D 16x16 map)
#pragma unroll
  for (int tm = 0; tm < 2; tm++) {
#pragma unroll
    for (int tn = 0; tn < 4; tn++) {
      const int mg = blockM + wm + tm * 16 + 8 * h;
      const int ng = blockN + wn + tn * 16 + l16;
      v8f a = acc[tm][tn];
      if constexpr (TRANS) {
        if constexpr (sizeof(OutT) == 2) {
          v8bf sv;
#pragma unroll
          for (int j = 0; j < 8; j++) sv[j] = (bf16)a[j];
          *reinterpret_cast<v8bf*>(&C[(size_t)ng * ldc + mg]) = sv;  // 16B contig
        } else {
#pragma unroll
          for (int j = 0; j < 8; j++) C[(size_t)ng * ldc + mg + j] = (OutT)a[j];
        }
      } else {
#pragma unroll
        for (int j = 0; j < 8; j++) C[(size_t)(mg + j) * ldc + ng] = (OutT)a[j];
      }
    }
  }
}

// ---------------- launcher --------------------------------------------------
extern "C" void kernel_launch(void* const* d_in, const int* in_sizes, int n_in,
                              void* d_out, int out_size, void* d_ws, size_t ws_size,
                              hipStream_t stream) {
  (void)in_sizes; (void)n_in; (void)out_size; (void)ws_size;
  const float* inputs = (const float*)d_in[0];
  const float* w      = (const float*)d_in[1];
  const float* w_q    = (const float*)d_in[2];
  const float* w_k    = (const float*)d_in[3];
  const float* w_v    = (const float*)d_in[4];
  float* out = (float*)d_out;
  char* ws = (char*)d_ws;

  // Workspace layout (bytes). qT reuses the inputs_bf16 region (consumed by GEMM-a).
  bf16*  inb  = (bf16*)(ws + 0ull);           // 32MB  [N,D] bf16 (later: qT [D,N])
  bf16*  qT   = inb;
  bf16*  aB   = (bf16*)(ws + 33554432ull);    // 32MB  a   [N,D]
  bf16*  kT   = (bf16*)(ws + 67108864ull);    // 32MB  k^T [D,N]
  bf16*  vB   = (bf16*)(ws + 100663296ull);   // 32MB  v   [N,D]
  bf16*  wT   = (bf16*)(ws + 134217728ull);   // 8MB   W^T
  bf16*  wqT  = (bf16*)(ws + 142606336ull);   // 8MB
  bf16*  wkT  = (bf16*)(ws + 150994944ull);   // 8MB
  bf16*  wvT  = (bf16*)(ws + 159383552ull);   // 8MB
  float* kq   = (float*)(ws + 167772160ull);  // 16MB  k_q [D,D] fp32
  bf16*  kqhT = (bf16*)(ws + 184549376ull);   // 8MB   softmax(k_q)^T bf16

  // 1) conversions
  cvt_f32_bf16<<<(NROWS * DDIM) / 1024, 256, 0, stream>>>(inputs, inb);
  dim3 tb(32, 8), tg(DDIM / 32, DDIM / 32);
  cvt_f32_bf16_T<<<tg, tb, 0, stream>>>(w,   wT,  DDIM);
  cvt_f32_bf16_T<<<tg, tb, 0, stream>>>(w_q, wqT, DDIM);
  cvt_f32_bf16_T<<<tg, tb, 0, stream>>>(w_k, wkT, DDIM);
  cvt_f32_bf16_T<<<tg, tb, 0, stream>>>(w_v, wvT, DDIM);

  // 2) GEMM chain
  dim3 g1(DDIM / BN, NROWS / BM);  // (16, 64)
  gemm_bf16_wmma<bf16, false><<<g1, 256, 0, stream>>>(inb, wT,  aB, NROWS, DDIM, DDIM, DDIM, DDIM, DDIM);
  gemm_bf16_wmma<bf16, true ><<<g1, 256, 0, stream>>>(aB,  wqT, qT, NROWS, DDIM, DDIM, DDIM, DDIM, NROWS);
  gemm_bf16_wmma<bf16, true ><<<g1, 256, 0, stream>>>(aB,  wkT, kT, NROWS, DDIM, DDIM, DDIM, DDIM, NROWS);
  gemm_bf16_wmma<bf16, false><<<g1, 256, 0, stream>>>(aB,  wvT, vB, NROWS, DDIM, DDIM, DDIM, DDIM, DDIM);

  // 3) k_q = k^T q : A = k^T [D, N] (lda=N), Bt = q^T [D, N] (ldb=N)
  dim3 g2(DDIM / BN, DDIM / BM);   // (16, 16)
  gemm_bf16_wmma<float, false><<<g2, 256, 0, stream>>>(kT, qT, kq, DDIM, DDIM, NROWS, NROWS, NROWS, DDIM);

  // 4) softmax over columns of each row, write transposed bf16
  softmax_rows_T<<<DDIM, 256, 0, stream>>>(kq, kqhT, DDIM);

  // 5) out = v @ k_q_hat : Bt = k_q_hat^T
  gemm_bf16_wmma<float, false><<<g1, 256, 0, stream>>>(vB, kqhT, out, NROWS, DDIM, DDIM, DDIM, DDIM, DDIM);
}
